// CrossAttention_14070312861921
// MI455X (gfx1250) — matrix operand order repaired and checked by reading, not compiled, not run on hardware
//
#include <hip/hip_runtime.h>
#include <cmath>

// ---------------------------------------------------------------------------
// CDNA5 (gfx1250) cross-attention: wave32, WMMA f32_16x16x32_f16,
// async-to-LDS (ASYNCcnt) double-buffered K/V staging in flash attention.
// ---------------------------------------------------------------------------

typedef __attribute__((ext_vector_type(16))) _Float16 v16h;
typedef __attribute__((ext_vector_type(8)))  _Float16 v8h;
typedef __attribute__((ext_vector_type(8)))  float    v8f;

#define DEV __device__ __forceinline__

#define BATCH  4
#define NSEQ   2048   // Nq == Nk
#define DMODEL 512
#define NHEAD  8
#define DHEAD  64
#define ROWS   (BATCH * NSEQ)   // 8192

union HF { v16h v; v8h h[2]; };

// A-fragment (16x32 f16, row-major source, row already selected by lane&15):
// lanes 0-15:  e[0..7]=K0..7,  e[8..15]=K16..23
// lanes 16-31: e[0..7]=K8..15, e[8..15]=K24..31
DEV v16h load_afrag(const _Float16* row_kk, int hi) {
    HF f;
    const _Float16* p = row_kk + (hi ? 8 : 0);
    f.h[0] = *(const v8h*)(p);
    f.h[1] = *(const v8h*)(p + 16);
    return f.v;
}

// B-fragment (32x16 f16) where B[k][n] is contiguous in k for fixed n
// (source row n holds the K-run). lanes 0-15: K0..7,K16..23; lanes 16-31: K8..15,K24..31.
DEV v16h load_bfrag(const _Float16* row_kk, int hi) {
    HF f;
    const _Float16* p = row_kk + (hi ? 8 : 0);
    f.h[0] = *(const v8h*)(p);
    f.h[1] = *(const v8h*)(p + 16);
    return f.v;
}

DEV v8f wmma16x16x32(v16h a, v16h b, v8f c) {
    return __builtin_amdgcn_wmma_f32_16x16x32_f16(
        /*neg_a=*/false, a, /*neg_b=*/false, b,
        /*c_mod=*/(short)0, c, /*reuse_a=*/false, /*reuse_b=*/false);
}

// gfx1250 async copy: 16B global -> LDS, tracked by ASYNCcnt.
DEV void async_g2lds_b128(unsigned int lds_off, const void* gaddr) {
    asm volatile("global_load_async_to_lds_b128 %0, %1, off"
                 :: "v"(lds_off),
                    "v"((unsigned long long)(uintptr_t)gaddr)
                 : "memory");
}

// ---------------------------------------------------------------------------
// Elementwise f32 -> f16 cast (weights + raw V)
// ---------------------------------------------------------------------------
__global__ void cast_f32_to_f16(const float* __restrict__ x,
                                _Float16* __restrict__ y, int n) {
    int i = blockIdx.x * blockDim.x + threadIdx.x;
    if (i < n) y[i] = (_Float16)x[i];
}

// ---------------------------------------------------------------------------
// LayerNorm over D=512, one wave per row, output f16
// ---------------------------------------------------------------------------
__global__ void ln_rows_f16(const float* __restrict__ X, const float* __restrict__ g,
                            const float* __restrict__ bta, _Float16* __restrict__ Y,
                            int rows) {
    int wave = threadIdx.x >> 5, lane = threadIdx.x & 31;
    int row = blockIdx.x * (blockDim.x >> 5) + wave;
    if (row >= rows) return;
    const float* x = X + (size_t)row * DMODEL;
    float v[16]; float s = 0.f, sq = 0.f;
#pragma unroll
    for (int i = 0; i < 16; ++i) {
        float t = x[i * 32 + lane];
        v[i] = t; s += t; sq += t * t;
    }
#pragma unroll
    for (int m = 1; m <= 16; m <<= 1) {
        s  += __shfl_xor(s,  m, 32);
        sq += __shfl_xor(sq, m, 32);
    }
    float mean = s * (1.f / DMODEL);
    float var  = sq * (1.f / DMODEL) - mean * mean;
    float rs   = rsqrtf(var + 1e-5f);
    _Float16* y = Y + (size_t)row * DMODEL;
#pragma unroll
    for (int i = 0; i < 16; ++i) {
        int c = i * 32 + lane;
        y[c] = (_Float16)((v[i] - mean) * rs * g[c] + bta[c]);
    }
}

// Same, but emits f32 (residual path) and f16 (GEMM operand)
__global__ void ln_rows_dual(const float* __restrict__ X, const float* __restrict__ g,
                             const float* __restrict__ bta, float* __restrict__ Y32,
                             _Float16* __restrict__ Y16, int rows) {
    int wave = threadIdx.x >> 5, lane = threadIdx.x & 31;
    int row = blockIdx.x * (blockDim.x >> 5) + wave;
    if (row >= rows) return;
    const float* x = X + (size_t)row * DMODEL;
    float v[16]; float s = 0.f, sq = 0.f;
#pragma unroll
    for (int i = 0; i < 16; ++i) {
        float t = x[i * 32 + lane];
        v[i] = t; s += t; sq += t * t;
    }
#pragma unroll
    for (int m = 1; m <= 16; m <<= 1) {
        s  += __shfl_xor(s,  m, 32);
        sq += __shfl_xor(sq, m, 32);
    }
    float mean = s * (1.f / DMODEL);
    float var  = sq * (1.f / DMODEL) - mean * mean;
    float rs   = rsqrtf(var + 1e-5f);
    float*    y32 = Y32 + (size_t)row * DMODEL;
    _Float16* y16 = Y16 + (size_t)row * DMODEL;
#pragma unroll
    for (int i = 0; i < 16; ++i) {
        int c = i * 32 + lane;
        float r = (v[i] - mean) * rs * g[c] + bta[c];
        y32[c] = r;
        y16[c] = (_Float16)r;
    }
}

// ---------------------------------------------------------------------------
// Head-scatter store of one 16x16 C-fragment (f32 acc -> f16 memory).
// ---------------------------------------------------------------------------
DEV void store_head_tile(v8f acc, int m0, int n0, int hi, int lane,
                         _Float16* __restrict__ Y, int transpose_out) {
    int nloc = n0 + (lane & 15);
    int h = nloc >> 6, d = nloc & 63;
#pragma unroll
    for (int r = 0; r < 8; ++r) {
        int mloc = m0 + (hi ? r + 8 : r);
        int bb = mloc >> 11, rr = mloc & (NSEQ - 1);
        size_t idx = transpose_out
            ? (((size_t)bb * NHEAD + h) * DHEAD + d) * NSEQ + rr
            : (((size_t)bb * NHEAD + h) * NSEQ + rr) * DHEAD + d;
        Y[idx] = (_Float16)acc[r];
    }
}

// ---------------------------------------------------------------------------
// Y = X @ W^T projections, scattered to per-head layouts.
//   X: [8192, 512] f16 row-major, W: [512, 512] f16 row-major (B[k][n]=W[n][k])
//   transpose_out==0 : Y[b][h][row][d]   ([B,H,N,64],  for q / k)
//   transpose_out==1 : Y[b][h][d][row]   ([B,H,64,N],  for v -> P@V B-operand)
// One wave per 32x32 tile (2x2 WMMA sub-tiles): 64 B of loads per output.
// K loop unrolled x4 only: ~16 fragment loads in flight keeps the wave under
// ~200 VGPRs (4-5 waves/SIMD) instead of spilling into the >256 VGPR file
// with single-wave occupancy, while still giving 16 WMMAs per body to hide
// L2 latency behind the s_wait_loadcnt staircase.
// ---------------------------------------------------------------------------
__global__ __launch_bounds__(128)
void gemm_xwt_head(const _Float16* __restrict__ X, const _Float16* __restrict__ W,
                   _Float16* __restrict__ Y, int transpose_out) {
    int wave = threadIdx.x >> 5, lane = threadIdx.x & 31;
    int hi = (lane >> 4) & 1;
    int tm = blockIdx.y * 4 + wave;   // 0..255 (32 rows each)
    int tn = blockIdx.x;              // 0..15  (32 cols each)
    int m0 = tm * 32, n0 = tn * 32;
    const _Float16* arow0 = X + (size_t)(m0 + (lane & 15)) * DMODEL;
    const _Float16* arow1 = arow0 + (size_t)16 * DMODEL;
    const _Float16* brow0 = W + (size_t)(n0 + (lane & 15)) * DMODEL;
    const _Float16* brow1 = brow0 + (size_t)16 * DMODEL;
    v8f a00 = {}, a01 = {}, a10 = {}, a11 = {};
#pragma unroll 4
    for (int kk = 0; kk < DMODEL; kk += 32) {
        v16h fa0 = load_afrag(arow0 + kk, hi);
        v16h fa1 = load_afrag(arow1 + kk, hi);
        v16h fb0 = load_bfrag(brow0 + kk, hi);
        v16h fb1 = load_bfrag(brow1 + kk, hi);
        a00 = wmma16x16x32(fa0, fb0, a00);
        a01 = wmma16x16x32(fa0, fb1, a01);
        a10 = wmma16x16x32(fa1, fb0, a10);
        a11 = wmma16x16x32(fa1, fb1, a11);
    }
    store_head_tile(a00, m0,      n0,      hi, lane, Y, transpose_out);
    store_head_tile(a01, m0,      n0 + 16, hi, lane, Y, transpose_out);
    store_head_tile(a10, m0 + 16, n0,      hi, lane, Y, transpose_out);
    store_head_tile(a11, m0 + 16, n0 + 16, hi, lane, Y, transpose_out);
}

// ---------------------------------------------------------------------------
// Flash attention. Block = 4 waves sharing (b,h); each wave owns a 16-row
// q tile. K/V kv-tiles (64 wide) are staged into LDS once per block with
// async-to-LDS copies (ASYNCcnt), double-buffered: the next tile streams in
// while the current one is consumed from LDS.
//   Qh/Kh: [B,H,N,64] f16; Vt: [B,H,64,N] f16; O: [B,N,512] f32.
// ---------------------------------------------------------------------------
__global__ __launch_bounds__(128)
void flash_attn(const _Float16* __restrict__ Qh, const _Float16* __restrict__ Kh,
                const _Float16* __restrict__ Vt, float* __restrict__ O) {
    __shared__ _Float16 Kt[2][64 * 64];     // [kv][dh]  8 KB per buffer
    __shared__ _Float16 Vs[2][64 * 64];     // [dh][kv]  8 KB per buffer
    __shared__ _Float16 Pbuf[4][16 * 64];   // per-wave P tile (C->A relayout)
    int tid = threadIdx.x;
    int wave = tid >> 5, lane = tid & 31;
    int hi = (lane >> 4) & 1;
    int bh = blockIdx.x;                    // 0..31
    int b = bh >> 3, h = bh & 7;
    int qt = blockIdx.y * 4 + wave;         // 0..127
    const float SCALE = 0.044194173824159216f;   // 1/sqrt(512) (TEMPERATURE)

    const _Float16* qrow = Qh + ((size_t)bh * NSEQ + qt * 16 + (lane & 15)) * DHEAD;
    v16h aq0 = load_afrag(qrow + 0,  hi);
    v16h aq1 = load_afrag(qrow + 32, hi);

    const _Float16* Kbase = Kh + (size_t)bh * NSEQ * DHEAD;   // contiguous kv tiles
    const _Float16* Vbase = Vt + (size_t)bh * DHEAD * NSEQ;   // 64 rows, stride NSEQ

    // 8 async instrs / wave / tile: 4 for K (contiguous 8KB), 4 for V (strided rows)
    auto issue_tiles = [&](int buf, int kv) {
        const char* gk = (const char*)(Kbase + (size_t)kv * DHEAD);
        const char* gv = (const char*)(Vbase + kv);
#pragma unroll
        for (int i = 0; i < 4; ++i) {
            int c = tid + 128 * i;                       // 0..511 x 16B = 8KB
            unsigned int lk = (unsigned int)(uintptr_t)(&Kt[buf][0]) + (unsigned int)(c * 16);
            async_g2lds_b128(lk, gk + (size_t)c * 16);
        }
#pragma unroll
        for (int i = 0; i < 4; ++i) {
            int c = tid + 128 * i;
            int row = c >> 3, col = (c & 7) * 16;        // 64 rows x 128B
            unsigned int lv = (unsigned int)(uintptr_t)(&Vs[buf][0]) + (unsigned int)(row * 128 + col);
            async_g2lds_b128(lv, gv + (size_t)row * (NSEQ * 2) + col);
        }
    };

    float rowmax[8], rowsum[8];
    v8f oacc[4];
#pragma unroll
    for (int r = 0; r < 8; ++r) { rowmax[r] = -1e30f; rowsum[r] = 0.f; }
#pragma unroll
    for (int d = 0; d < 4; ++d) { v8f z = {}; oacc[d] = z; }

    _Float16* Pl = &Pbuf[wave][0];

    issue_tiles(0, 0);   // prologue prefetch

    for (int it = 0; it < NSEQ / 64; ++it) {
        int cur = it & 1;
        if (it + 1 < NSEQ / 64) {
            issue_tiles(cur ^ 1, (it + 1) * 64);
            asm volatile("s_wait_asynccnt 8" ::: "memory");  // current tile landed
        } else {
            asm volatile("s_wait_asynccnt 0" ::: "memory");
        }
        __syncthreads();                                     // all waves' fills visible

        // ---- S tile (16 x 64) = Q Kt from LDS ------------------------------
        v8f s[4];
#pragma unroll
        for (int j = 0; j < 4; ++j) {
            const _Float16* krow = &Kt[cur][(j * 16 + (lane & 15)) * 64];
            v16h bk0 = load_bfrag(krow + 0,  hi);
            v16h bk1 = load_bfrag(krow + 32, hi);
            v8f acc = {};
            acc = wmma16x16x32(aq0, bk0, acc);
            acc = wmma16x16x32(aq1, bk1, acc);
#pragma unroll
            for (int r = 0; r < 8; ++r) {
                float xx = acc[r] * SCALE;
                acc[r] = fminf(fmaxf(xx, -10000.f), 10000.f);
            }
            s[j] = acc;
        }
        // ---- online softmax (lanes 0-15 -> rows r, lanes 16-31 -> rows r+8)
        float tmax[8];
#pragma unroll
        for (int r = 0; r < 8; ++r) {
            float m = s[0][r];
            m = fmaxf(m, s[1][r]); m = fmaxf(m, s[2][r]); m = fmaxf(m, s[3][r]);
            tmax[r] = m;
        }
#pragma unroll
        for (int msk = 1; msk <= 8; msk <<= 1)
#pragma unroll
            for (int r = 0; r < 8; ++r)
                tmax[r] = fmaxf(tmax[r], __shfl_xor(tmax[r], msk, 32));

        float alpha[8];
#pragma unroll
        for (int r = 0; r < 8; ++r) {
            float nm = fmaxf(rowmax[r], tmax[r]);
            alpha[r] = __expf(rowmax[r] - nm);
            rowmax[r] = nm;
        }
        float tsum[8];
#pragma unroll
        for (int r = 0; r < 8; ++r) tsum[r] = 0.f;
#pragma unroll
        for (int j = 0; j < 4; ++j)
#pragma unroll
            for (int r = 0; r < 8; ++r) {
                float p = __expf(s[j][r] - rowmax[r]);
                s[j][r] = p;
                tsum[r] += p;
            }
#pragma unroll
        for (int msk = 1; msk <= 8; msk <<= 1)
#pragma unroll
            for (int r = 0; r < 8; ++r)
                tsum[r] += __shfl_xor(tsum[r], msk, 32);
#pragma unroll
        for (int r = 0; r < 8; ++r) rowsum[r] = rowsum[r] * alpha[r] + tsum[r];
#pragma unroll
        for (int d = 0; d < 4; ++d)
#pragma unroll
            for (int r = 0; r < 8; ++r) oacc[d][r] *= alpha[r];

        // ---- C-layout -> A-layout via private LDS slice --------------------
#pragma unroll
        for (int j = 0; j < 4; ++j)
#pragma unroll
            for (int r = 0; r < 8; ++r) {
                int m = hi ? (r + 8) : r;
                int n = j * 16 + (lane & 15);
                Pl[m * 64 + n] = (_Float16)s[j][r];
            }
        asm volatile("s_wait_dscnt 0" ::: "memory");   // own stores -> own loads

        const _Float16* prow = Pl + (lane & 15) * 64;
        v16h ap0 = load_afrag(prow + 0,  hi);
        v16h ap1 = load_afrag(prow + 32, hi);

        // ---- O += P @ V from LDS (Vs row = head-dim, contiguous in kv) -----
#pragma unroll
        for (int d = 0; d < 4; ++d) {
            const _Float16* vrow = &Vs[cur][(d * 16 + (lane & 15)) * 64];
            v16h bv0 = load_bfrag(vrow + 0,  hi);
            v16h bv1 = load_bfrag(vrow + 32, hi);
            oacc[d] = wmma16x16x32(ap0, bv0, oacc[d]);
            oacc[d] = wmma16x16x32(ap1, bv1, oacc[d]);
        }
        __syncthreads();   // all reads done before this buffer is refilled
    }

    // ---- normalize + store [B, Nq, 512] f32 --------------------------------
#pragma unroll
    for (int d = 0; d < 4; ++d)
#pragma unroll
        for (int r = 0; r < 8; ++r) {
            int m = qt * 16 + (hi ? r + 8 : r);
            int n = h * DHEAD + d * 16 + (lane & 15);
            O[((size_t)b * NSEQ + m) * DMODEL + n] = oacc[d][r] / rowsum[r];
        }
}

// ---------------------------------------------------------------------------
// Out = R + GELU_exact(X @ W^T); X f16 [8192,512], W f16 [512,512], R/Out f32
// 32x32 tile per wave, K loop unrolled x4 (same reasoning as projection GEMM).
// ---------------------------------------------------------------------------
__global__ __launch_bounds__(128)
void gemm_xwt_gelu(const _Float16* __restrict__ X, const _Float16* __restrict__ W,
                   const float* __restrict__ R, float* __restrict__ Out) {
    int wave = threadIdx.x >> 5, lane = threadIdx.x & 31;
    int hi = (lane >> 4) & 1;
    int tm = blockIdx.y * 4 + wave;
    int tn = blockIdx.x;
    int m0 = tm * 32, n0 = tn * 32;
    const _Float16* arow0 = X + (size_t)(m0 + (lane & 15)) * DMODEL;
    const _Float16* arow1 = arow0 + (size_t)16 * DMODEL;
    const _Float16* brow0 = W + (size_t)(n0 + (lane & 15)) * DMODEL;
    const _Float16* brow1 = brow0 + (size_t)16 * DMODEL;
    v8f a00 = {}, a01 = {}, a10 = {}, a11 = {};
#pragma unroll 4
    for (int kk = 0; kk < DMODEL; kk += 32) {
        v16h fa0 = load_afrag(arow0 + kk, hi);
        v16h fa1 = load_afrag(arow1 + kk, hi);
        v16h fb0 = load_bfrag(brow0 + kk, hi);
        v16h fb1 = load_bfrag(brow1 + kk, hi);
        a00 = wmma16x16x32(fa0, fb0, a00);
        a01 = wmma16x16x32(fa0, fb1, a01);
        a10 = wmma16x16x32(fa1, fb0, a10);
        a11 = wmma16x16x32(fa1, fb1, a11);
    }
    v8f accs[2][2] = { { a00, a01 }, { a10, a11 } };
#pragma unroll
    for (int ms = 0; ms < 2; ++ms)
#pragma unroll
        for (int ns = 0; ns < 2; ++ns) {
            int nloc = n0 + ns * 16 + (lane & 15);
#pragma unroll
            for (int r = 0; r < 8; ++r) {
                int mloc = m0 + ms * 16 + (hi ? r + 8 : r);
                float gx = accs[ms][ns][r];
                float ge = 0.5f * gx * (1.f + erff(gx * 0.70710678118654752f));
                size_t idx = (size_t)mloc * DMODEL + nloc;
                Out[idx] = R[idx] + ge;
            }
        }
}

// ---------------------------------------------------------------------------
// Host-side orchestration (graph-capture safe: kernel launches only)
// ---------------------------------------------------------------------------
extern "C" void kernel_launch(void* const* d_in, const int* in_sizes, int n_in,
                              void* d_out, int out_size, void* d_ws, size_t ws_size,
                              hipStream_t stream) {
    (void)in_sizes; (void)n_in; (void)out_size; (void)ws_size;
    const float* Q   = (const float*)d_in[0];
    const float* K   = (const float*)d_in[1];
    const float* V   = (const float*)d_in[2];
    const float* Wq  = (const float*)d_in[3];
    const float* Wk  = (const float*)d_in[4];
    const float* Wv  = (const float*)d_in[5];
    const float* Wo  = (const float*)d_in[6];
    const float* gqk = (const float*)d_in[7];
    const float* bqk = (const float*)d_in[8];
    const float* go  = (const float*)d_in[9];
    const float* bo  = (const float*)d_in[10];

    char* ws = (char*)d_ws;
    const size_t KB = 1024, MB = 1024 * 1024;
    // weights (persistent)
    _Float16* Wq16 = (_Float16*)(ws + 0 * 512 * KB);
    _Float16* Wk16 = (_Float16*)(ws + 1 * 512 * KB);
    _Float16* Wv16 = (_Float16*)(ws + 2 * 512 * KB);
    _Float16* Wo16 = (_Float16*)(ws + 3 * 512 * KB);
    // region A [2MB,26MB): LN outputs, later reused for attention output
    _Float16* Qn16 = (_Float16*)(ws + 2 * MB);
    _Float16* Kn16 = (_Float16*)(ws + 10 * MB);
    _Float16* V16  = (_Float16*)(ws + 18 * MB);
    float*    Obuf = (float*)   (ws + 2 * MB);    // reuses Qn16/Kn16 (dead by then)
    // region B [26MB,50MB): projected q/k/v, later reused for LN(O) outputs
    _Float16* qh   = (_Float16*)(ws + 26 * MB);
    _Float16* kh   = (_Float16*)(ws + 34 * MB);
    _Float16* vt   = (_Float16*)(ws + 42 * MB);
    float*    On32 = (float*)   (ws + 26 * MB);   // reuses qh/kh (dead by then)
    _Float16* On16 = (_Float16*)(ws + 42 * MB);   // reuses vt

    const int NW = DMODEL * DMODEL;               // 262144
    const int NV = ROWS * DMODEL;                 // 4194304

    // 1) casts
    cast_f32_to_f16<<<(NW + 255) / 256, 256, 0, stream>>>(Wq, Wq16, NW);
    cast_f32_to_f16<<<(NW + 255) / 256, 256, 0, stream>>>(Wk, Wk16, NW);
    cast_f32_to_f16<<<(NW + 255) / 256, 256, 0, stream>>>(Wv, Wv16, NW);
    cast_f32_to_f16<<<(NW + 255) / 256, 256, 0, stream>>>(Wo, Wo16, NW);
    cast_f32_to_f16<<<(NV + 255) / 256, 256, 0, stream>>>(V, V16, NV);

    // 2) LayerNorm Q, K (shared params)
    ln_rows_f16<<<ROWS / 8, 256, 0, stream>>>(Q, gqk, bqk, Qn16, ROWS);
    ln_rows_f16<<<ROWS / 8, 256, 0, stream>>>(K, gqk, bqk, Kn16, ROWS);

    // 3) projections (WMMA, 32x32 tiles/wave)
    dim3 gg(DMODEL / 32, ROWS / 32 / 4), bg(128);
    gemm_xwt_head<<<gg, bg, 0, stream>>>(Qn16, Wq16, qh, 0);
    gemm_xwt_head<<<gg, bg, 0, stream>>>(Kn16, Wk16, kh, 0);
    gemm_xwt_head<<<gg, bg, 0, stream>>>(V16,  Wv16, vt, 1);

    // 4) flash attention (WMMA + online softmax + async LDS staging)
    flash_attn<<<dim3(BATCH * NHEAD, NSEQ / 16 / 4), 128, 0, stream>>>(qh, kh, vt, Obuf);

    // 5) LayerNorm(O) -> f32 residual + f16 operand
    ln_rows_dual<<<ROWS / 8, 256, 0, stream>>>(Obuf, go, bo, On32, On16, ROWS);

    // 6) residual + GELU(fc_o) (WMMA)
    gemm_xwt_gelu<<<gg, bg, 0, stream>>>(On16, Wo16, On32, (float*)d_out);
}
